// ProtGraphTransformer_3547642987149
// MI455X (gfx1250) — matrix operand order repaired and checked by reading, not compiled, not run on hardware
//
#include <hip/hip_runtime.h>
#include <hip/hip_bf16.h>

typedef __attribute__((ext_vector_type(16))) _Float16 v16h;
typedef __attribute__((ext_vector_type(4)))  _Float16 v4h;
typedef __attribute__((ext_vector_type(8)))  float    v8f;

#define TMB 32      // rows per block (2 m-tiles of 16)
#define KI  64      // k per pipeline stage (2 WMMA k-chunks)
#define HDIM 128

static __device__ __forceinline__ unsigned enc_f32(float f) {
  unsigned b = __float_as_uint(f);
  return (b & 0x80000000u) ? ~b : (b | 0x80000000u);
}
static __device__ __forceinline__ float dec_f32(unsigned k) {
  return (k & 0x80000000u) ? __uint_as_float(k & 0x7FFFFFFFu)
                           : __uint_as_float(~k);
}

// CDNA5 async global->LDS copy (ASYNCcnt path, cdna5_isa 15.18.3 op 98)
static __device__ __forceinline__ void async_copy_b128(unsigned lds_off,
                                                       const void* gaddr) {
  asm volatile("global_load_async_to_lds_b128 %0, %1, off"
               :: "v"(lds_off), "v"(gaddr) : "memory");
}
static __device__ __forceinline__ void wait_async0() {
  asm volatile("s_wait_asynccnt 0" ::: "memory");
}

// ---------------------------------------------------------------------------
// Pack fp32 weight W[Kin x 128] into f16, chunk-major fragment order:
//   Wh[kc*4096 + col*32 + kk] = W[(kc*32+kk)*128 + col]
// Each 32x128 k-chunk is a contiguous 8KB block laid out [col][k]; a 64-k
// pipeline stage is therefore a contiguous 16KB block, async-copyable
// straight into LDS with per-lane fragments contiguous.
// ---------------------------------------------------------------------------
__global__ void pgt_pack_w(const float* __restrict__ W,
                           _Float16* __restrict__ Wh, int Kin) {
  int o = blockIdx.x * 256 + threadIdx.x;
  if (o >= Kin * HDIM) return;
  int kc  = o >> 12;       // / 4096
  int rem = o & 4095;
  int col = rem >> 5;
  int kk  = rem & 31;
  Wh[o] = (_Float16)W[((kc << 5) + kk) * HDIM + col];
}

// ---------------------------------------------------------------------------
// Gathered GEMM:  C[M x 128] = act( A' @ W + bias )
// A' rows composed of up to 3 fp32 segments of width 128:
//   segment s element = As[s][ (Is[s] ? Is[s][row] : row)*128 + col ]
// 256 threads = 8 waves; block computes 32 rows x 128 cols; wave w owns cols
// [16w,16w+16) and both 16-row m-tiles.
// Double-buffered pipeline, 64-k stages: per stage a wave issues 4 WMMAs;
// the next stage's async B copy + A global loads are in flight underneath.
// ---------------------------------------------------------------------------
__global__ __launch_bounds__(256) void pgt_gemm_gather_wmma(
    int M, int Kin,
    const float* __restrict__ A0, const int* __restrict__ I0,
    const float* __restrict__ A1, const int* __restrict__ I1,
    const float* __restrict__ A2, const int* __restrict__ I2,
    const _Float16* __restrict__ Wh, const float* __restrict__ bias,
    float* __restrict__ C, int relu)
{
  // A fragments: [buf][sub(2)][tile(2) x lane(32) x elem(16)]
  __shared__ alignas(32) _Float16 As[2][2][2 * 32 * 16];
  // B stage: [buf][sub*4096 + col*32 + kk]  (== packed Wh stage layout)
  __shared__ alignas(32) _Float16 Bs[2][HDIM * KI];

  const int tid  = threadIdx.x;
  const int row0 = blockIdx.x * TMB;

  // per-thread gathered row indices (8 threads per row, r = tid>>3)
  const int r = tid >> 3;            // 0..31
  int gr = row0 + r;
  if (gr >= M) gr = M - 1;           // clamp; stores guarded later
  const int mr0 = I0 ? I0[gr] : gr;
  const int mr1 = (Kin > 128) ? (I1 ? I1[gr] : gr) : 0;
  const int mr2 = (Kin > 256) ? (I2 ? I2[gr] : gr) : 0;

  const int wave = tid >> 5;
  const int lane = tid & 31;
  const int n0   = wave * 16;
  const int n    = lane & 15;
  const int half = lane >> 4;

  const unsigned bs_off0 = (unsigned)(size_t)(void*)&Bs[0][0];
  const unsigned bs_off1 = (unsigned)(size_t)(void*)&Bs[1][0];

  // A staging geometry (constant per thread): 4 consecutive cols -> one lane
  const int c0    = (tid & 7) * 4;                      // col-in-chunk 0..28
  const int tA    = r >> 4;                             // m-tile
  const int laneA = (r & 15) + ((c0 >> 4) << 4);        // dest lane
  const int jA    = c0 & 15;                            // dest elem
  const int aoff  = tA * 512 + laneA * 16 + jA;         // within [sub] block

  auto issueB = [&](int it, int buf) {
    const _Float16* gsrc = Wh + (size_t)it * (HDIM * KI) + tid * 32;
    unsigned dst = (buf ? bs_off1 : bs_off0) + (unsigned)tid * 64u;
    async_copy_b128(dst,       (const void*)(gsrc));
    async_copy_b128(dst + 16u, (const void*)(gsrc + 8));
    async_copy_b128(dst + 32u, (const void*)(gsrc + 16));
    async_copy_b128(dst + 48u, (const void*)(gsrc + 24));
  };
  auto loadA = [&](int it, int sub) -> float4 {
    const int seg = it >> 1;                 // 2 stages per 128-wide segment
    const float* A  = (seg == 0) ? A0 : (seg == 1) ? A1 : A2;
    const int    mr = (seg == 0) ? mr0 : (seg == 1) ? mr1 : mr2;
    const int colbase = (it & 1) * KI + sub * 32;
    return *(const float4*)(A + (long)mr * HDIM + colbase + c0);
  };
  auto storeA = [&](int buf, int sub, float4 f) {
    v4h h4 = { (_Float16)f.x, (_Float16)f.y, (_Float16)f.z, (_Float16)f.w };
    *(v4h*)(&As[buf][sub][aoff]) = h4;
  };

  v8f acc0 = {};
  v8f acc1 = {};
  const int nIter = Kin / KI;

  // prologue: stage 0 -> buffer 0
  issueB(0, 0);
  {
    float4 p0 = loadA(0, 0);
    float4 p1 = loadA(0, 1);
    storeA(0, 0, p0);
    storeA(0, 1, p1);
  }
  wait_async0();
  __syncthreads();

  for (int it = 0; it < nIter; ++it) {
    const int cur = it & 1, nxt = cur ^ 1;
    const bool more = (it + 1) < nIter;
    float4 q0 = {}, q1 = {};
    if (more) {
      issueB(it + 1, nxt);          // async DMA underneath the WMMAs
      q0 = loadA(it + 1, 0);        // global loads in flight
      q1 = loadA(it + 1, 1);
    }

    const _Float16* Ac = &As[cur][0][0];
    const _Float16* Bc = &Bs[cur][0];
    v16h a00 = *(const v16h*)(Ac + lane * 16);          // sub0, tile0
    v16h a10 = *(const v16h*)(Ac + 512  + lane * 16);   // sub0, tile1
    v16h a01 = *(const v16h*)(Ac + 1024 + lane * 16);   // sub1, tile0
    v16h a11 = *(const v16h*)(Ac + 1536 + lane * 16);   // sub1, tile1
    v16h b0  = *(const v16h*)(Bc + (n0 + n) * 32 + half * 16);
    v16h b1  = *(const v16h*)(Bc + 4096 + (n0 + n) * 32 + half * 16);

    acc0 = __builtin_amdgcn_wmma_f32_16x16x32_f16(false, a00, false, b0,
                                                  (short)0, acc0, false, false);
    acc1 = __builtin_amdgcn_wmma_f32_16x16x32_f16(false, a10, false, b0,
                                                  (short)0, acc1, false, false);
    acc0 = __builtin_amdgcn_wmma_f32_16x16x32_f16(false, a01, false, b1,
                                                  (short)0, acc0, false, false);
    acc1 = __builtin_amdgcn_wmma_f32_16x16x32_f16(false, a11, false, b1,
                                                  (short)0, acc1, false, false);

    if (more) {
      storeA(nxt, 0, q0);           // cvt+store after compute issued
      storeA(nxt, 1, q1);
      wait_async0();
    }
    __syncthreads();
  }

  // ---- epilogue: bias + optional relu ----
  const float bv = bias[n0 + n];
#pragma unroll
  for (int v = 0; v < 8; ++v) {
    int m  = v + 8 * half;
    int g0 = row0 + m;
    int g1 = row0 + 16 + m;
    if (g0 < M) {
      float x = acc0[v] + bv;
      if (relu) x = fmaxf(x, 0.f);
      C[(long)g0 * HDIM + n0 + n] = x;
    }
    if (g1 < M) {
      float x = acc1[v] + bv;
      if (relu) x = fmaxf(x, 0.f);
      C[(long)g1 * HDIM + n0 + n] = x;
    }
  }
}

// ---------------------------------------------------------------------------
// Elementwise / attention kernels
// ---------------------------------------------------------------------------
__global__ void pgt_fill_u32(unsigned* __restrict__ p, unsigned v, long n) {
  long t = (long)blockIdx.x * blockDim.x + threadIdx.x;
  if (t < n) p[t] = v;
}

__global__ void pgt_attn_score(long E_, const int* __restrict__ dst,
                               const float* __restrict__ Q,
                               const float* __restrict__ K,
                               float* __restrict__ s) {
  long t = (long)blockIdx.x * blockDim.x + threadIdx.x;
  if (t >= E_ * 4) return;
  long e = t >> 2;
  int  h = (int)(t & 3);
  const float* q = Q + (long)dst[e] * HDIM + h * 32;
  const float* k = K + e * HDIM + h * 32;
  float acc = 0.f;
#pragma unroll
  for (int d = 0; d < 32; ++d) acc += q[d] * k[d];
  s[t] = acc * 0.17677669529663687f;   // 1/sqrt(32)
}

__global__ void pgt_seg_max(long E_, const int* __restrict__ dst,
                            const float* __restrict__ s,
                            unsigned* __restrict__ smax_enc) {
  long t = (long)blockIdx.x * blockDim.x + threadIdx.x;
  if (t >= E_ * 4) return;
  long e = t >> 2;
  int  h = (int)(t & 3);
  atomicMax(&smax_enc[(long)dst[e] * 4 + h], enc_f32(s[t]));
}

__global__ void pgt_exp_denom(long E_, const int* __restrict__ dst,
                              const unsigned* __restrict__ smax_enc,
                              float* __restrict__ s,
                              float* __restrict__ denom) {
  long t = (long)blockIdx.x * blockDim.x + threadIdx.x;
  if (t >= E_ * 4) return;
  long e = t >> 2;
  int  h = (int)(t & 3);
  long di = (long)dst[e] * 4 + h;
  float av = __expf(s[t] - dec_f32(smax_enc[di]));
  s[t] = av;
  atomicAdd(&denom[di], av);
}

__global__ void pgt_aggregate(long E_, const int* __restrict__ dst,
                              const float* __restrict__ V,
                              const float* __restrict__ a,
                              float* __restrict__ hagg) {
  long t = (long)blockIdx.x * blockDim.x + threadIdx.x;
  if (t >= E_ * HDIM) return;
  long e = t >> 7;
  int  c = (int)(t & 127);
  int  h = c >> 5;
  float val = V[e * HDIM + c] * a[e * 4 + h];
  atomicAdd(&hagg[(long)dst[e] * HDIM + c], val);
}

__global__ void pgt_normalize(long N_, const float* __restrict__ denom,
                              float* __restrict__ hagg) {
  long t = (long)blockIdx.x * blockDim.x + threadIdx.x;
  if (t >= N_ * HDIM) return;
  float dn = denom[(t >> 7) * 4 + ((t & 127) >> 5)];
  hagg[t] = hagg[t] / (dn > 0.f ? dn : 1.f);
}

// out = LayerNorm(base + delta) * g + b, one wave32 per 128-wide row
__global__ __launch_bounds__(256) void pgt_ln_residual(
    long M, const float* __restrict__ base, const float* __restrict__ delta,
    const float* __restrict__ g, const float* __restrict__ b,
    float* __restrict__ out)
{
  long row = (long)blockIdx.x * 8 + (threadIdx.x >> 5);
  int lane = threadIdx.x & 31;
  if (row >= M) return;
  const float* xr = base  + row * HDIM;
  const float* dr = delta + row * HDIM;
  float v[4], s1 = 0.f, s2 = 0.f;
#pragma unroll
  for (int k = 0; k < 4; ++k) {
    float t = xr[lane + 32 * k] + dr[lane + 32 * k];
    v[k] = t; s1 += t; s2 += t * t;
  }
#pragma unroll
  for (int off = 16; off > 0; off >>= 1) {
    s1 += __shfl_xor(s1, off, 32);
    s2 += __shfl_xor(s2, off, 32);
  }
  float mean = s1 * (1.f / 128.f);
  float var  = s2 * (1.f / 128.f) - mean * mean;
  float inv  = rsqrtf(var + 1e-5f);
#pragma unroll
  for (int k = 0; k < 4; ++k) {
    int c = lane + 32 * k;
    out[row * HDIM + c] = (v[k] - mean) * inv * g[c] + b[c];
  }
}

// ---------------------------------------------------------------------------
// Host side
// ---------------------------------------------------------------------------
static inline long cdivl(long a, long b) { return (a + b - 1) / b; }

extern "C" void kernel_launch(void* const* d_in, const int* in_sizes, int n_in,
                              void* d_out, int out_size, void* d_ws, size_t ws_size,
                              hipStream_t stream) {
  const float* h   = (const float*)d_in[0];
  const float* e   = (const float*)d_in[1];
  const int*   src = (const int*)d_in[2];
  const int*   dst = (const int*)d_in[3];
  const float* Wq  = (const float*)d_in[4];
  const float* bq  = (const float*)d_in[5];
  const float* Wk  = (const float*)d_in[6];
  const float* bk  = (const float*)d_in[7];
  const float* Wv  = (const float*)d_in[8];
  const float* bv  = (const float*)d_in[9];
  const float* Wn1 = (const float*)d_in[10];
  const float* bn1 = (const float*)d_in[11];
  const float* Wn2 = (const float*)d_in[12];
  const float* bn2 = (const float*)d_in[13];
  const float* We1 = (const float*)d_in[14];
  const float* be1 = (const float*)d_in[15];
  const float* We2 = (const float*)d_in[16];
  const float* be2 = (const float*)d_in[17];
  const float* gh  = (const float*)d_in[18];
  const float* bh  = (const float*)d_in[19];
  const float* ge  = (const float*)d_in[20];
  const float* be  = (const float*)d_in[21];

  const long N_ = in_sizes[0] / HDIM;   // 50000
  const long E_ = in_sizes[1] / HDIM;   // 800000

  // workspace layout, 256B aligned
  char* ws = (char*)d_ws;
  auto align256 = [](size_t x) { return (x + 255) & ~(size_t)255; };
  size_t off = 0;
  _Float16* Whq = (_Float16*)(ws + off); off = align256(off + (size_t)128 * HDIM * 2);
  _Float16* Whk = (_Float16*)(ws + off); off = align256(off + (size_t)256 * HDIM * 2);
  _Float16* Whv = (_Float16*)(ws + off); off = align256(off + (size_t)256 * HDIM * 2);
  _Float16* Whn1= (_Float16*)(ws + off); off = align256(off + (size_t)128 * HDIM * 2);
  _Float16* Whn2= (_Float16*)(ws + off); off = align256(off + (size_t)128 * HDIM * 2);
  _Float16* Whe1= (_Float16*)(ws + off); off = align256(off + (size_t)384 * HDIM * 2);
  _Float16* Whe2= (_Float16*)(ws + off); off = align256(off + (size_t)128 * HDIM * 2);
  float*    Qf    = (float*)(ws + off); off = align256(off + (size_t)N_ * HDIM * 4);
  float*    Kf    = (float*)(ws + off); off = align256(off + (size_t)E_ * HDIM * 4);
  float*    Vf    = (float*)(ws + off); off = align256(off + (size_t)E_ * HDIM * 4);
  float*    sbuf  = (float*)(ws + off); off = align256(off + (size_t)E_ * 4 * 4);
  unsigned* smaxe = (unsigned*)(ws + off); off = align256(off + (size_t)N_ * 4 * 4);
  float*    denom = (float*)(ws + off); off = align256(off + (size_t)N_ * 4 * 4);
  float*    hagg  = (float*)(ws + off); off = align256(off + (size_t)N_ * HDIM * 4);
  float*    t1    = (float*)(ws + off); off = align256(off + (size_t)E_ * HDIM * 4);
  (void)ws_size; (void)n_in; (void)out_size;

  float* h_out = (float*)d_out;                     // [N,128]
  float* e_out = (float*)d_out + (size_t)N_ * HDIM; // [E,128]

  // pack weights to f16 fragment-order chunks (tiny)
  pgt_pack_w<<<cdivl(128 * HDIM, 256), 256, 0, stream>>>(Wq,  Whq,  128);
  pgt_pack_w<<<cdivl(256 * HDIM, 256), 256, 0, stream>>>(Wk,  Whk,  256);
  pgt_pack_w<<<cdivl(256 * HDIM, 256), 256, 0, stream>>>(Wv,  Whv,  256);
  pgt_pack_w<<<cdivl(128 * HDIM, 256), 256, 0, stream>>>(Wn1, Whn1, 128);
  pgt_pack_w<<<cdivl(128 * HDIM, 256), 256, 0, stream>>>(Wn2, Whn2, 128);
  pgt_pack_w<<<cdivl(384 * HDIM, 256), 256, 0, stream>>>(We1, Whe1, 384);
  pgt_pack_w<<<cdivl(128 * HDIM, 256), 256, 0, stream>>>(We2, Whe2, 128);

  // init accumulation buffers (ws is poisoned; must re-init every call)
  pgt_fill_u32<<<cdivl(N_ * 4, 256), 256, 0, stream>>>(smaxe, 0x007FFFFFu, N_ * 4); // enc(-inf)
  pgt_fill_u32<<<cdivl(N_ * 4, 256), 256, 0, stream>>>((unsigned*)denom, 0u, N_ * 4);
  pgt_fill_u32<<<cdivl(N_ * HDIM, 256), 256, 0, stream>>>((unsigned*)hagg, 0u, N_ * HDIM);

  // Q = h @ Wq + bq
  pgt_gemm_gather_wmma<<<cdivl(N_, TMB), 256, 0, stream>>>(
      (int)N_, 128, h, nullptr, nullptr, nullptr, nullptr, nullptr,
      Whq, bq, Qf, 0);
  // K = [h[src] | e] @ Wk + bk
  pgt_gemm_gather_wmma<<<cdivl(E_, TMB), 256, 0, stream>>>(
      (int)E_, 256, h, src, e, nullptr, nullptr, nullptr,
      Whk, bk, Kf, 0);
  // V = [h[src] | e] @ Wv + bv
  pgt_gemm_gather_wmma<<<cdivl(E_, TMB), 256, 0, stream>>>(
      (int)E_, 256, h, src, e, nullptr, nullptr, nullptr,
      Whv, bv, Vf, 0);

  // attention scores + segment softmax + weighted scatter
  pgt_attn_score<<<cdivl(E_ * 4, 256), 256, 0, stream>>>(E_, dst, Qf, Kf, sbuf);
  pgt_seg_max   <<<cdivl(E_ * 4, 256), 256, 0, stream>>>(E_, dst, sbuf, smaxe);
  pgt_exp_denom <<<cdivl(E_ * 4, 256), 256, 0, stream>>>(E_, dst, smaxe, sbuf, denom);
  pgt_aggregate <<<cdivl(E_ * HDIM, 256), 256, 0, stream>>>(E_, dst, Vf, sbuf, hagg);
  pgt_normalize <<<cdivl(N_ * HDIM, 256), 256, 0, stream>>>(N_, denom, hagg);

  // node MLP: h_new = relu(relu(hagg@Wn1+bn1)@Wn2+bn2); h_out = LN(h + h_new)
  pgt_gemm_gather_wmma<<<cdivl(N_, TMB), 256, 0, stream>>>(
      (int)N_, 128, hagg, nullptr, nullptr, nullptr, nullptr, nullptr,
      Whn1, bn1, t1, 1);
  pgt_gemm_gather_wmma<<<cdivl(N_, TMB), 256, 0, stream>>>(
      (int)N_, 128, t1, nullptr, nullptr, nullptr, nullptr, nullptr,
      Whn2, bn2, Qf, 1);
  pgt_ln_residual<<<cdivl(N_, 8), 256, 0, stream>>>(N_, h, Qf, gh, bh, h_out);

  // edge MLP: me = [h_out[src] | h_out[dst] | e]
  pgt_gemm_gather_wmma<<<cdivl(E_, TMB), 256, 0, stream>>>(
      (int)E_, 384, h_out, src, h_out, dst, e, nullptr,
      Whe1, be1, t1, 1);
  pgt_gemm_gather_wmma<<<cdivl(E_, TMB), 256, 0, stream>>>(
      (int)E_, 128, t1, nullptr, nullptr, nullptr, nullptr, nullptr,
      Whe2, be2, Kf, 1);
  pgt_ln_residual<<<cdivl(E_, 8), 256, 0, stream>>>(E_, e, Kf, ge, be, e_out);
}
